// SpanBERT_CRF_87196426043986
// MI455X (gfx1250) — matrix operand order repaired and checked by reading, not compiled, not run on hardware
//
#include <hip/hip_runtime.h>
#include <hip/hip_bf16.h>

// ---------------------------------------------------------------------------
// SpanBERT + CRF forward for MI455X (gfx1250).
// GEMMs: v_wmma_f32_16x16x32_f16, double-buffered LDS tiles fed exclusively by
// global_load_async_to_lds_b128 (ASYNCcnt DMA). All B operands pre-transposed
// (weights via tiled convert-transpose, V via tiled f16 transpose) so staging
// is straight 16B async copies. FF1 fuses exact GELU into the epilogue.
// ---------------------------------------------------------------------------

typedef __attribute__((ext_vector_type(16))) _Float16 v16h;
typedef __attribute__((ext_vector_type(8)))  float    v8f;
typedef _Float16 h16;

#define BB  8
#define SS  512
#define HH  768
#define LL  12
#define NH  12
#define DH  64
#define FF  3072
#define NEGBIAS (-1000000000.0f)

// ----------------------------- WMMA GEMM -----------------------------------
#define BM 128
#define BN 64
#define BK 32

union HalfVec { float4 f4[2]; v16h h; };

// Async DMA: 16 bytes global -> LDS, tracked by ASYNCcnt.
// Low 32 bits of a generic LDS pointer are the LDS byte offset.
__device__ __forceinline__ void async_copy_b128(const h16* g, h16* l)
{
    const unsigned lds = (unsigned)(unsigned long long)(void*)l;
    const unsigned long long ga = (unsigned long long)g;
    asm volatile("global_load_async_to_lds_b128 %0, %1, off"
                 :: "v"(lds), "v"(ga) : "memory");
}

__device__ __forceinline__ void wait_async0()
{
    asm volatile("s_wait_asynccnt 0x0" ::: "memory");
}

// Stage one (BM x BK) A tile and one (BN x BK) B^T tile; 3 async b128/thread.
__device__ __forceinline__ void stage_tile(const h16* __restrict__ A,
                                           const h16* __restrict__ Bt,
                                           h16* As, h16* Bs,
                                           int m0, int n0, int k0, int K,
                                           int lda, int ldb, int tid)
{
    {
        const int row = tid >> 1;
        const int kh  = (tid & 1) * 16;
        const h16* src = A + (long long)(m0 + row) * lda + k0 + kh;
        if (k0 + BK < K)  // keep L2 warm for the tile after next
            __builtin_prefetch(src + BK, 0, 1);
        async_copy_b128(src,     &As[row * BK + kh]);
        async_copy_b128(src + 8, &As[row * BK + kh + 8]);
    }
    {
        const int n  = tid >> 2;
        const int kh = (tid & 3) * 8;
        const h16* src = Bt + (long long)(n0 + n) * ldb + k0 + kh;
        if (k0 + BK < K)
            __builtin_prefetch(src + BK, 0, 1);
        async_copy_b128(src, &Bs[n * BK + kh]);
    }
}

// C[M,N] = alpha * A[M,K] x B^T[N,K] (+bias), EPI: 0=linear, 1=exact GELU.
template<int EPI, bool WRITE32, bool WRITE16>
__global__ __launch_bounds__(256)
void wmma_gemm(const h16* __restrict__ A, const h16* __restrict__ Bt,
               const float* __restrict__ bias,
               float* __restrict__ C32, h16* __restrict__ C16,
               int M, int N, int K, int lda, int ldb, int ldc,
               int zdiv,
               long long sA1, long long sA2,
               long long sB1, long long sB2,
               long long sC1, long long sC2,
               float alpha)
{
    __shared__ __align__(16) h16 As[2][BM * BK];   // [row][k], stride BK
    __shared__ __align__(16) h16 Bs[2][BN * BK];   // [col][k], stride BK

    const int z  = blockIdx.z;
    const int b1 = z / zdiv;
    const int b2 = z % zdiv;
    A  += b1 * sA1 + b2 * sA2;
    Bt += b1 * sB1 + b2 * sB2;
    const long long cOff = b1 * sC1 + b2 * sC2;

    const int m0   = blockIdx.y * BM;
    const int n0   = blockIdx.x * BN;
    const int tid  = threadIdx.x;
    const int lane = tid & 31;
    const int wave = tid >> 5;              // 0..7
    const int waveM = (wave & 3) * 32;      // 4 waves along M
    const int waveN = (wave >> 2) * 32;     // 2 waves along N

    v8f acc[2][2] = {};

    // ---- prologue: stage first K tile into buffer 0 ----
    stage_tile(A, Bt, As[0], Bs[0], m0, n0, 0, K, lda, ldb, tid);
    wait_async0();
    __syncthreads();

    int buf = 0;
    for (int k0 = 0; k0 < K; k0 += BK) {
        // ---- issue async DMA for the next K tile into the other buffer ----
        if (k0 + BK < K)
            stage_tile(A, Bt, As[buf ^ 1], Bs[buf ^ 1],
                       m0, n0, k0 + BK, K, lda, ldb, tid);

        // ---- load fragments per ISA VGPR layouts ----
        // A 16x32: lane row = m + (lane&15); halves: k = (h/8)*16 + (lane>>4)*8 + h%8
        HalfVec af[2], bf[2];
        const int arow  = waveM + (lane & 15);
        const int base8 = (lane >> 4) * 8;
        #pragma unroll
        for (int fi = 0; fi < 2; ++fi) {
            const h16* ap = &As[buf][(arow + fi * 16) * BK];
            af[fi].f4[0] = *reinterpret_cast<const float4*>(ap + base8);
            af[fi].f4[1] = *reinterpret_cast<const float4*>(ap + 16 + base8);
        }
        // B 32x16: lane col = n + (lane&15); halves: k = (h/8)*16 + (lane>>4)*8 + h%8
        const int bcol = waveN + (lane & 15);
        const int kb   = (lane >> 4) * 8;
        #pragma unroll
        for (int fj = 0; fj < 2; ++fj) {
            const h16* bp = &Bs[buf][(bcol + fj * 16) * BK + kb];
            bf[fj].f4[0] = *reinterpret_cast<const float4*>(bp);
            bf[fj].f4[1] = *reinterpret_cast<const float4*>(bp + 16);
        }
        #pragma unroll
        for (int fi = 0; fi < 2; ++fi)
            #pragma unroll
            for (int fj = 0; fj < 2; ++fj)
                acc[fi][fj] = __builtin_amdgcn_wmma_f32_16x16x32_f16(
                    false, af[fi].h, false, bf[fj].h,
                    (short)0, acc[fi][fj], false, false);

        // next-tile DMA must be complete workgroup-wide before it is consumed
        wait_async0();
        __syncthreads();
        buf ^= 1;
    }

    // ---- epilogue: C/D layout: n = tile + (lane&15); m = tile + r + (lane>>4)*8
    #pragma unroll
    for (int fi = 0; fi < 2; ++fi) {
        #pragma unroll
        for (int fj = 0; fj < 2; ++fj) {
            const int n = n0 + waveN + fj * 16 + (lane & 15);
            const float bv = bias ? bias[n] : 0.0f;
            const int mbase = m0 + waveM + fi * 16 + (lane >> 4) * 8;
            #pragma unroll
            for (int r = 0; r < 8; ++r) {
                float val = acc[fi][fj][r] * alpha + bv;
                if constexpr (EPI == 1)   // exact GELU
                    val = 0.5f * val * (1.0f + erff(val * 0.70710678118654752f));
                const long long idx = cOff + (long long)(mbase + r) * ldc + n;
                if constexpr (WRITE32) C32[idx] = val;
                if constexpr (WRITE16) C16[idx] = (h16)val;
            }
        }
    }
}

// ------------------------- transpose kernels --------------------------------

// W: KxN f32 (row-major)  ->  Wt: NxK f16 (row-major). Grid (N/32, K/32), blk (32,8).
__global__ __launch_bounds__(256)
void convert_transpose_kernel(const float* __restrict__ W, h16* __restrict__ Wt,
                              int K, int N)
{
    __shared__ float tile[32][33];
    const int k0 = blockIdx.y * 32;
    const int n0 = blockIdx.x * 32;
    const int tx = threadIdx.x, ty = threadIdx.y;
    #pragma unroll
    for (int j = 0; j < 32; j += 8)
        tile[ty + j][tx] = W[(long long)(k0 + ty + j) * N + n0 + tx];
    __syncthreads();
    #pragma unroll
    for (int j = 0; j < 32; j += 8)
        Wt[(long long)(n0 + ty + j) * K + k0 + tx] = (h16)tile[tx][ty + j];
}

// v16 [B,S,NH,DH] f16 -> vT [B,NH,DH,S] f16. Grid (S/32, DH/32, B*NH), blk (32,8).
__global__ __launch_bounds__(256)
void transpose_v_kernel(const h16* __restrict__ vin, h16* __restrict__ vout)
{
    __shared__ h16 tile[32][33];
    const int bh = blockIdx.z;               // b*NH + h
    const int b  = bh / NH, hd = bh % NH;
    const int s0 = blockIdx.x * 32;
    const int d0 = blockIdx.y * 32;
    const int tx = threadIdx.x, ty = threadIdx.y;
    #pragma unroll
    for (int j = 0; j < 32; j += 8)
        tile[ty + j][tx] = vin[((long long)(b * SS + s0 + ty + j) * NH + hd) * DH + d0 + tx];
    __syncthreads();
    #pragma unroll
    for (int j = 0; j < 32; j += 8)
        vout[((long long)bh * DH + d0 + ty + j) * SS + s0 + tx] = tile[tx][ty + j];
}

// ------------------------- elementwise kernels ------------------------------

__device__ __forceinline__ void row_ln_768(float loc[3], const float* g, const float* b,
                                           float* y32, h16* y16, float* red)
{
    const int tid = threadIdx.x;
    float s = loc[0] + loc[1] + loc[2];
    red[tid] = s; __syncthreads();
    for (int st = 128; st; st >>= 1) { if (tid < st) red[tid] += red[tid + st]; __syncthreads(); }
    const float mean = red[0] / (float)HH; __syncthreads();
    float v = 0.f;
    #pragma unroll
    for (int c = 0; c < 3; ++c) { float t = loc[c] - mean; v += t * t; }
    red[tid] = v; __syncthreads();
    for (int st = 128; st; st >>= 1) { if (tid < st) red[tid] += red[tid + st]; __syncthreads(); }
    const float inv = rsqrtf(red[0] / (float)HH + 1e-12f);
    #pragma unroll
    for (int c = 0; c < 3; ++c) {
        const int i = tid + c * 256;
        const float o = (loc[c] - mean) * inv * g[i] + b[i];
        y32[i] = o;
        y16[i] = (h16)o;
    }
}

__global__ __launch_bounds__(256)
void embed_ln_kernel(const int* __restrict__ ids, const int* __restrict__ tts,
                     const float* __restrict__ we, const float* __restrict__ pe,
                     const float* __restrict__ te,
                     const float* __restrict__ g, const float* __restrict__ b,
                     float* __restrict__ x32, h16* __restrict__ x16)
{
    __shared__ float red[256];
    const int row = blockIdx.x;           // b*S + s
    const int s   = row & (SS - 1);
    const int id  = ids[row];
    const int tt  = tts[row];
    float loc[3];
    #pragma unroll
    for (int c = 0; c < 3; ++c) {
        const int i = threadIdx.x + c * 256;
        loc[c] = we[(long long)id * HH + i] + pe[(long long)s * HH + i] + te[(long long)tt * HH + i];
    }
    row_ln_768(loc, g, b, x32 + (long long)row * HH, x16 + (long long)row * HH, red);
}

__global__ __launch_bounds__(256)
void add_ln_kernel(float* __restrict__ x32, const float* __restrict__ d,
                   const float* __restrict__ g, const float* __restrict__ b,
                   h16* __restrict__ x16)
{
    __shared__ float red[256];
    const long long row = blockIdx.x;
    float loc[3];
    #pragma unroll
    for (int c = 0; c < 3; ++c) {
        const int i = threadIdx.x + c * 256;
        loc[c] = x32[row * HH + i] + d[row * HH + i];
    }
    row_ln_768(loc, g, b, x32 + row * HH, x16 + row * HH, red);
}

__global__ __launch_bounds__(256)
void softmax_kernel(const float* __restrict__ scores, h16* __restrict__ P,
                    const int* __restrict__ mask)
{
    __shared__ float red[256];
    const long long row = blockIdx.x;          // (b*NH + h)*S + q
    const int b = (int)(row / ((long long)NH * SS));
    const float* src = scores + row * SS;
    h16* dst = P + row * SS;
    const int* mrow = mask + (long long)b * SS;
    const int tid = threadIdx.x;

    float vals[2];
    float mx = -3.0e38f;
    #pragma unroll
    for (int c = 0; c < 2; ++c) {
        const int j = tid + c * 256;
        const float v = src[j] + (1.0f - (float)mrow[j]) * NEGBIAS;
        vals[c] = v;
        mx = fmaxf(mx, v);
    }
    red[tid] = mx; __syncthreads();
    for (int st = 128; st; st >>= 1) { if (tid < st) red[tid] = fmaxf(red[tid], red[tid + st]); __syncthreads(); }
    mx = red[0]; __syncthreads();
    float sum = 0.f;
    #pragma unroll
    for (int c = 0; c < 2; ++c) { vals[c] = __expf(vals[c] - mx); sum += vals[c]; }
    red[tid] = sum; __syncthreads();
    for (int st = 128; st; st >>= 1) { if (tid < st) red[tid] += red[tid + st]; __syncthreads(); }
    const float inv = 1.0f / red[0];
    #pragma unroll
    for (int c = 0; c < 2; ++c) dst[tid + c * 256] = (h16)(vals[c] * inv);
}

__global__ __launch_bounds__(256)
void qa_kernel(const float* __restrict__ x, const float* __restrict__ W,
               const float* __restrict__ bia, float* __restrict__ logits)
{
    __shared__ float r0[256], r1[256];
    const long long row = blockIdx.x;
    const int tid = threadIdx.x;
    float s0 = 0.f, s1 = 0.f;
    for (int i = tid; i < HH; i += 256) {
        const float v = x[row * HH + i];
        s0 += v * W[i * 2 + 0];
        s1 += v * W[i * 2 + 1];
    }
    r0[tid] = s0; r1[tid] = s1; __syncthreads();
    for (int st = 128; st; st >>= 1) {
        if (tid < st) { r0[tid] += r0[tid + st]; r1[tid] += r1[tid + st]; }
        __syncthreads();
    }
    if (tid == 0) {
        logits[row * 2 + 0] = r0[0] + bia[0];
        logits[row * 2 + 1] = r1[0] + bia[1];
    }
}

__device__ __forceinline__ float lse2(float a, float b)
{
    const float m = fmaxf(a, b);
    return m + logf(expf(a - m) + expf(b - m));
}

__global__ void crf_kernel(const float* __restrict__ logits, const int* __restrict__ mask,
                           const int* __restrict__ sp, const int* __restrict__ ep,
                           const float* __restrict__ startT, const float* __restrict__ endT,
                           const float* __restrict__ trans, float* __restrict__ out)
{
    __shared__ float acc[32];
    const int b = threadIdx.x;
    float llh = 0.f;
    if (b < BB) {
        const float* em = logits + (long long)b * SS * 2;
        const int* mk = mask + (long long)b * SS;
        const int s0 = sp[b], e0 = ep[b];
        const bool valid = (s0 <= e0) && (e0 < SS) && (s0 >= 0);
        const float t00 = trans[0], t01 = trans[1], t10 = trans[2], t11 = trans[3];

        // numerator
        int prev = (valid && 0 >= s0 && 0 <= e0) ? 1 : 0;
        float num = startT[prev] + em[prev];
        int msum = mk[0];
        for (int t = 1; t < SS; ++t) {
            const int tg = (valid && t >= s0 && t <= e0) ? 1 : 0;
            const float m = (float)mk[t];
            num += (trans[prev * 2 + tg] + em[t * 2 + tg]) * m;
            msum += mk[t];
            prev = tg;
        }
        const int last = msum - 1;
        const int ltag = (valid && last >= s0 && last <= e0) ? 1 : 0;
        num += endT[ltag];

        // denominator: 2-state forward algorithm
        float a0 = startT[0] + em[0];
        float a1 = startT[1] + em[1];
        for (int t = 1; t < SS; ++t) {
            const float e0v = em[t * 2], e1v = em[t * 2 + 1];
            const float n0 = lse2(a0 + t00, a1 + t10) + e0v;
            const float n1 = lse2(a0 + t01, a1 + t11) + e1v;
            if (mk[t]) { a0 = n0; a1 = n1; }
        }
        const float den = lse2(a0 + endT[0], a1 + endT[1]);
        llh = num - den;
    }
    acc[threadIdx.x] = llh;
    __syncthreads();
    if (threadIdx.x == 0) {
        float s = 0.f;
        for (int i = 0; i < BB; ++i) s += acc[i];
        out[0] = -s;
    }
}

// ------------------------------ host side -----------------------------------

static inline char* carve(char*& p, size_t bytes)
{
    char* r = p;
    p += (bytes + 255) & ~(size_t)255;
    return r;
}

extern "C" void kernel_launch(void* const* d_in, const int* in_sizes, int n_in,
                              void* d_out, int out_size, void* d_ws, size_t ws_size,
                              hipStream_t stream)
{
    (void)in_sizes; (void)n_in; (void)out_size; (void)ws_size;

    const int*   input_ids = (const int*)  d_in[0];
    const int*   attn_mask = (const int*)  d_in[1];
    const int*   type_ids  = (const int*)  d_in[2];
    const int*   start_pos = (const int*)  d_in[3];
    const int*   end_pos   = (const int*)  d_in[4];
    const float* word_emb  = (const float*)d_in[5];
    const float* pos_emb   = (const float*)d_in[6];
    const float* type_emb  = (const float*)d_in[7];
    const float* emb_g     = (const float*)d_in[8];
    const float* emb_b     = (const float*)d_in[9];
    const float* Wq = (const float*)d_in[10]; const float* bq = (const float*)d_in[11];
    const float* Wk = (const float*)d_in[12]; const float* bk = (const float*)d_in[13];
    const float* Wv = (const float*)d_in[14]; const float* bv = (const float*)d_in[15];
    const float* Wo = (const float*)d_in[16]; const float* bo = (const float*)d_in[17];
    const float* g1 = (const float*)d_in[18]; const float* b1 = (const float*)d_in[19];
    const float* W1 = (const float*)d_in[20]; const float* c1 = (const float*)d_in[21];
    const float* W2 = (const float*)d_in[22]; const float* c2 = (const float*)d_in[23];
    const float* g2 = (const float*)d_in[24]; const float* b2 = (const float*)d_in[25];
    const float* qaW = (const float*)d_in[26]; const float* qab = (const float*)d_in[27];
    const float* crf_s = (const float*)d_in[28];
    const float* crf_e = (const float*)d_in[29];
    const float* crf_t = (const float*)d_in[30];

    const long long T = (long long)BB * SS;            // 4096 tokens

    char* p = (char*)d_ws;
    float* x32    = (float*)carve(p, T * HH * 4);
    h16*   x16    = (h16*)  carve(p, T * HH * 2);
    h16*   wt16   = (h16*)  carve(p, (size_t)HH * FF * 2);   // transposed weight staging
    h16*   q16    = (h16*)  carve(p, T * HH * 2);
    h16*   k16    = (h16*)  carve(p, T * HH * 2);
    h16*   v16    = (h16*)  carve(p, T * HH * 2);
    h16*   vT16   = (h16*)  carve(p, T * HH * 2);            // [B,NH,DH,S]
    float* scores = (float*)carve(p, (size_t)BB * NH * SS * SS * 4);
    h16*   P16    = (h16*)  carve(p, (size_t)BB * NH * SS * SS * 2);
    h16*   ctx16  = (h16*)  carve(p, T * HH * 2);
    float* tmp32  = (float*)carve(p, T * HH * 4);            // attn-out / ff2-out
    h16*   h16b   = (h16*)  carve(p, T * FF * 2);            // gelu(FF1) f16
    float* logits = (float*)carve(p, T * 2 * 4);

    // ---- embeddings + LN ----
    embed_ln_kernel<<<dim3((unsigned)T), 256, 0, stream>>>(
        input_ids, type_ids, word_emb, pos_emb, type_emb, emb_g, emb_b, x32, x16);

    const dim3 blkT(32, 8);
    const dim3 gTWH (HH / 32, HH / 32);           // transpose HxH weights
    const dim3 gTW1 (FF / 32, HH / 32);           // W1: 768x3072 -> 3072x768
    const dim3 gTW2 (HH / 32, FF / 32);           // W2: 3072x768 -> 768x3072
    const dim3 gTV  (SS / 32, DH / 32, BB * NH);  // V transpose

    const dim3 gProj(HH / BN, T / BM, 1);         // 12 x 32
    const dim3 gFF1 (FF / BN, T / BM, 1);         // 48 x 32
    const dim3 gQK  (SS / BN, SS / BM, BB * NH);
    const dim3 gPV  (DH / BN, SS / BM, BB * NH);

    for (int l = 0; l < LL; ++l) {
        const float* Wq_l = Wq + (long long)l * HH * HH;
        const float* Wk_l = Wk + (long long)l * HH * HH;
        const float* Wv_l = Wv + (long long)l * HH * HH;
        const float* Wo_l = Wo + (long long)l * HH * HH;
        const float* W1_l = W1 + (long long)l * HH * FF;
        const float* W2_l = W2 + (long long)l * FF * HH;

        // Q = x @ Wq + bq   (f16 out)
        convert_transpose_kernel<<<gTWH, blkT, 0, stream>>>(Wq_l, wt16, HH, HH);
        wmma_gemm<0, false, true><<<gProj, 256, 0, stream>>>(
            x16, wt16, bq + l * HH, nullptr, q16,
            (int)T, HH, HH, HH, HH, HH, 1, 0,0, 0,0, 0,0, 1.0f);
        // K
        convert_transpose_kernel<<<gTWH, blkT, 0, stream>>>(Wk_l, wt16, HH, HH);
        wmma_gemm<0, false, true><<<gProj, 256, 0, stream>>>(
            x16, wt16, bk + l * HH, nullptr, k16,
            (int)T, HH, HH, HH, HH, HH, 1, 0,0, 0,0, 0,0, 1.0f);
        // V
        convert_transpose_kernel<<<gTWH, blkT, 0, stream>>>(Wv_l, wt16, HH, HH);
        wmma_gemm<0, false, true><<<gProj, 256, 0, stream>>>(
            x16, wt16, bv + l * HH, nullptr, v16,
            (int)T, HH, HH, HH, HH, HH, 1, 0,0, 0,0, 0,0, 1.0f);
        // V -> V^T per (b,h): [B,NH,DH,S]
        transpose_v_kernel<<<gTV, blkT, 0, stream>>>(v16, vT16);

        // scores = Q @ K^T / 8   per (b, h)
        wmma_gemm<0, true, false><<<gQK, 256, 0, stream>>>(
            q16, k16, nullptr, scores, nullptr,
            SS, SS, DH, HH, HH, SS, NH,
            (long long)SS * HH, DH,
            (long long)SS * HH, DH,
            (long long)NH * SS * SS, (long long)SS * SS, 0.125f);

        // softmax (+mask bias), f16 probabilities
        softmax_kernel<<<dim3((unsigned)((long long)BB * NH * SS)), 256, 0, stream>>>(
            scores, P16, attn_mask);

        // ctx = P @ (V^T)^T  (f16 out, [B,S,NH,DH] == [B,S,H])
        wmma_gemm<0, false, true><<<gPV, 256, 0, stream>>>(
            P16, vT16, nullptr, nullptr, ctx16,
            SS, DH, SS, SS, SS, HH, NH,
            (long long)NH * SS * SS, (long long)SS * SS,
            (long long)NH * DH * SS, (long long)DH * SS,
            (long long)SS * HH, DH, 1.0f);

        // attn_out = ctx @ Wo + bo
        convert_transpose_kernel<<<gTWH, blkT, 0, stream>>>(Wo_l, wt16, HH, HH);
        wmma_gemm<0, true, false><<<gProj, 256, 0, stream>>>(
            ctx16, wt16, bo + l * HH, tmp32, nullptr,
            (int)T, HH, HH, HH, HH, HH, 1, 0,0, 0,0, 0,0, 1.0f);

        // x = LN(x + attn_out)
        add_ln_kernel<<<dim3((unsigned)T), 256, 0, stream>>>(x32, tmp32, g1 + l * HH, b1 + l * HH, x16);

        // h = gelu(x @ W1 + c1), fused epilogue, f16 out
        convert_transpose_kernel<<<gTW1, blkT, 0, stream>>>(W1_l, wt16, HH, FF);
        wmma_gemm<1, false, true><<<gFF1, 256, 0, stream>>>(
            x16, wt16, c1 + l * FF, nullptr, h16b,
            (int)T, FF, HH, HH, HH, FF, 1, 0,0, 0,0, 0,0, 1.0f);

        // ff2 = h @ W2 + c2
        convert_transpose_kernel<<<gTW2, blkT, 0, stream>>>(W2_l, wt16, FF, HH);
        wmma_gemm<0, true, false><<<gProj, 256, 0, stream>>>(
            h16b, wt16, c2 + l * HH, tmp32, nullptr,
            (int)T, HH, FF, FF, FF, HH, 1, 0,0, 0,0, 0,0, 1.0f);

        // x = LN(x + ff2)
        add_ln_kernel<<<dim3((unsigned)T), 256, 0, stream>>>(x32, tmp32, g2 + l * HH, b2 + l * HH, x16);
    }

    // QA head: logits = x @ qaW + qab   (N=2, do as a reduction)
    qa_kernel<<<dim3((unsigned)T), 256, 0, stream>>>(x32, qaW, qab, logits);

    // CRF negative log-likelihood -> scalar loss
    crf_kernel<<<1, 32, 0, stream>>>(logits, attn_mask, start_pos, end_pos,
                                     crf_s, crf_e, crf_t, (float*)d_out);
}
